// SparseAttentionEngine_56710748176488
// MI455X (gfx1250) — matrix operand order repaired and checked by reading, not compiled, not run on hardware
//
#include <hip/hip_runtime.h>

// ---------------------------------------------------------------------------
// Sparse top-k attention for MI455X (gfx1250, wave32, WMMA 16x16x32 bf16,
// Tensor Data Mover staging for GEMM weight tiles)
// ---------------------------------------------------------------------------

typedef __attribute__((ext_vector_type(16))) __bf16        v16bf;
typedef __attribute__((ext_vector_type(8)))  float         v8f;
typedef __attribute__((ext_vector_type(4)))  unsigned int  u32x4;
typedef __attribute__((ext_vector_type(8)))  int           i32x8;
typedef __attribute__((ext_vector_type(4)))  int           i32x4;

#define HIDDEN 768
#define HEADS  12
#define HDIM   64
#define SEQ    2048
#define BATCH  2
#define MTOT   (BATCH * SEQ)     // 4096 rows for all projections
#define KK_TOP 204               // int(2048 * 0.1)

__device__ __forceinline__ unsigned short f2bf(float f) {
    unsigned u = __float_as_uint(f);
    u += 0x7FFFu + ((u >> 16) & 1u);   // round to nearest even
    return (unsigned short)(u >> 16);
}

// order-preserving map: float -> uint32 (monotone for all finite values)
__device__ __forceinline__ unsigned ordkey(float f) {
    unsigned u = __float_as_uint(f);
    return (u & 0x80000000u) ? ~u : (u | 0x80000000u);
}

union BFrag {
    v16bf v;
    unsigned short u[16];
};

// fragment index helpers (ISA 7.12.2 bf16 layouts, wave32)
__device__ __forceinline__ int ka_idx(int j, int lane) {   // A 16x32: K index
    return ((j >> 3) << 4) + ((lane >> 4) << 3) + (((j >> 1) & 3) << 1) + (j & 1);
}
__device__ __forceinline__ int kb_idx(int j, int lane) {   // B 32x16: K index
    return ((j >> 3) << 4) + ((lane >> 4) << 3) + (j & 7);
}

// ---------------------------------------------------------------------------
// float32 -> bf16 elementwise convert
// ---------------------------------------------------------------------------
__global__ __launch_bounds__(256)
void cvt_f32_bf16(const float* __restrict__ src, unsigned short* __restrict__ dst, int n) {
    int i = blockIdx.x * blockDim.x + threadIdx.x;
    if (i < n) dst[i] = f2bf(src[i]);
}

// ---------------------------------------------------------------------------
// C[M,N] = A[M,K](bf16) x W[N,K]^T(bf16) + bias[N]
// 256 threads = 8 waves; workgroup tile 128x64; each wave a 32x32 subtile
// (2x2 WMMA register block, A/B fragments reused 2x each).
// W tile staged by the Tensor Data Mover with hardware row padding; A tile
// staged with 16B vector LDS copies.
// Dynamic LDS layout:  [0, 6144)  Bs 64 rows x 48 bf16 (pad stride 96B)
//                      [6144, 18432) As 128 rows x 48 bf16
// ---------------------------------------------------------------------------
template <bool OUT_BF16>
__global__ __launch_bounds__(256)
void gemm_nt_bf16(const unsigned short* __restrict__ A, const unsigned short* __restrict__ W,
                  const float* __restrict__ bias, void* __restrict__ Cout,
                  int M, int N, int K) {
    extern __shared__ char gsm[];
    unsigned short* Bs = (unsigned short*)gsm;            // offset 0 (TDM lds_addr)
    unsigned short* As = (unsigned short*)(gsm + 6144);

    const int t    = threadIdx.x;
    const int lane = t & 31;
    const int wave = t >> 5;          // 0..7
    const int wr   = wave >> 1;       // 0..3  (row block of 32)
    const int wc   = wave & 1;        // 0..1  (col block of 32)
    const int m0   = blockIdx.y * 128;
    const int n0   = blockIdx.x * 64;

    v8f acc00 = {}, acc01 = {}, acc10 = {}, acc11 = {};

    for (int k0 = 0; k0 < K; k0 += 32) {
        // ---- stage A tile 128x32 bf16 : 512 uint4, 2 per thread ----
        #pragma unroll
        for (int i = 0; i < 2; ++i) {
            int idx = t + i * 256;            // 0..511
            int row = idx >> 2, q = idx & 3;  // 4 x 16B per 64B row
            const uint4* ap = (const uint4*)(A + (size_t)(m0 + row) * K + k0);
            __builtin_prefetch(A + (size_t)(m0 + row) * K + k0 + 32, 0, 1);
            ((uint4*)&As[row * 48])[q] = ap[q];
        }
        // ---- stage W tile 64x32 bf16 via Tensor Data Mover (wave 0 only) ----
#if __has_builtin(__builtin_amdgcn_tensor_load_to_lds)
        if (t < 32) {
            unsigned long long gaddr =
                (unsigned long long)(const void*)(W + (size_t)n0 * K + k0);
            u32x4 g0;
            g0[0] = 1u;                                   // count=1 (valid descriptor)
            g0[1] = 0u;                                   // lds_addr = 0 (Bs)
            g0[2] = (unsigned)gaddr;                      // global_addr[31:0]
            g0[3] = (unsigned)((gaddr >> 32) & 0x01FFFFFFu) | (2u << 30); // type=2
            i32x8 g1;
            g1[0] = (int)((1u << 16)        // data_size = 2 bytes
                        | (1u << 20)        // pad_enable
                        | (3u << 22)        // pad_interval: 16 DWORDs (64B row)
                        | (7u << 25));      // pad_amount: 8 DWORDs (32B) -> 96B stride
            g1[1] = (int)(768u << 16);      // tensor_dim0 = 768 (bits 79:48 low half)
            g1[2] = (int)(768u << 16);      // tensor_dim1 = 768 (bits 111:80 low half)
            g1[3] = (int)(32u << 16);       // tile_dim0 = 32 (K extent, contiguous)
            g1[4] = 64;                     // tile_dim1 = 64 (N rows)
            g1[5] = 768;                    // tensor_dim0_stride = 768 elements
            g1[6] = 0;
            g1[7] = 0;
            i32x4 gz4 = {0, 0, 0, 0};
            i32x8 gz8 = {0, 0, 0, 0, 0, 0, 0, 0};
            __builtin_amdgcn_tensor_load_to_lds(g0, g1, gz4, gz4, gz8, 0);
            __builtin_amdgcn_s_wait_tensorcnt(0);
        }
#else
        {   // fallback: cooperative 16B vector staging
            int row = t >> 2, q = t & 3;
            const uint4* wp = (const uint4*)(W + (size_t)(n0 + row) * K + k0);
            if (row < 64) ((uint4*)&Bs[row * 48])[q] = wp[q];
        }
#endif
        __syncthreads();

        // ---- fragments + 2x2 WMMA block ----
        BFrag a0, a1, b0, b1;
        #pragma unroll
        for (int j = 0; j < 16; ++j) {
            int ka = ka_idx(j, lane);
            a0.u[j] = As[(wr * 32 + (lane & 15)) * 48 + ka];
            a1.u[j] = As[(wr * 32 + 16 + (lane & 15)) * 48 + ka];
            int kb = kb_idx(j, lane);
            b0.u[j] = Bs[(wc * 32 + (lane & 15)) * 48 + kb];
            b1.u[j] = Bs[(wc * 32 + 16 + (lane & 15)) * 48 + kb];
        }
        acc00 = __builtin_amdgcn_wmma_f32_16x16x32_bf16(false, a0.v, false, b0.v, (short)0, acc00, false, false);
        acc01 = __builtin_amdgcn_wmma_f32_16x16x32_bf16(false, a0.v, false, b1.v, (short)0, acc01, false, false);
        acc10 = __builtin_amdgcn_wmma_f32_16x16x32_bf16(false, a1.v, false, b0.v, (short)0, acc10, false, false);
        acc11 = __builtin_amdgcn_wmma_f32_16x16x32_bf16(false, a1.v, false, b1.v, (short)0, acc11, false, false);
        __syncthreads();
    }

    // ---- epilogue: C/D layout VGPR i -> M = i + 8*(lane>=16), N = lane&15 ----
    const int mB = m0 + wr * 32 + ((lane >> 4) << 3);
    const int nB = n0 + wc * 32 + (lane & 15);
    #pragma unroll
    for (int tm = 0; tm < 2; ++tm) {
        #pragma unroll
        for (int tn = 0; tn < 2; ++tn) {
            const v8f* acc = (tm == 0) ? (tn == 0 ? &acc00 : &acc01)
                                       : (tn == 0 ? &acc10 : &acc11);
            int n = nB + tn * 16;
            float bn = bias[n];
            #pragma unroll
            for (int i = 0; i < 8; ++i) {
                int m = mB + tm * 16 + i;
                float v = (*acc)[i] + bn;
                if (OUT_BF16)
                    ((unsigned short*)Cout)[(size_t)m * N + n] = f2bf(v);
                else
                    ((float*)Cout)[(size_t)m * N + n] = v;
            }
        }
    }
}

// ---------------------------------------------------------------------------
// Sparse attention core: one workgroup per (16-query tile, head, batch).
// Full 16x2048 score rows live in LDS (128KB of the 320KB WGP LDS) so the
// exact per-row top-204 threshold is found by bisection on float order-keys.
// Q/K/V are bf16; accumulation f32; output written bf16 for the final GEMM.
// ---------------------------------------------------------------------------
__global__ __launch_bounds__(128)
void sparse_attn(const unsigned short* __restrict__ Q, const unsigned short* __restrict__ Kc,
                 const unsigned short* __restrict__ V, unsigned short* __restrict__ O) {
    extern __shared__ char smem_raw[];
    float*          sc     = (float*)smem_raw;                              // [16][SEQ] f32
    unsigned short* qs     = (unsigned short*)(smem_raw + 16 * SEQ * 4);    // [16][72] bf16
    unsigned*       thrK   = (unsigned*)(smem_raw + 16 * SEQ * 4 + 16 * 72 * 2);
    float*          rowMax = (float*)(thrK + 16);
    float*          rowSum = (float*)(rowMax + 16);

    const int t    = threadIdx.x;
    const int lane = t & 31;
    const int wave = t >> 5;                    // 0..3
    const int q0   = blockIdx.x * 16;
    const int h    = blockIdx.y;
    const int b    = blockIdx.z;
    const size_t base = (size_t)b * SEQ;
    const int hoff = h * HDIM;

    // ---- stage Q tile (16x64 bf16, padded stride 72) : 1 uint4 per thread ----
    {
        int row = t >> 3, q = t & 7;            // 8 x 16B per 128B row
        const uint4* qp = (const uint4*)(Q + (base + q0 + row) * HIDDEN + hoff);
        ((uint4*)&qs[row * 72])[q] = qp[q];
    }
    __syncthreads();

    // ---- Phase 1: scores = Q K^T / sqrt(64), 16x16 WMMA tiles into LDS ----
    for (int kt = wave; kt < SEQ / 16; kt += 4) {
        int key0 = kt * 16;
        v8f acc = {};
        #pragma unroll
        for (int kk0 = 0; kk0 < HDIM; kk0 += 32) {
            BFrag a, bf;
            const unsigned short* kp = Kc + (base + key0 + (lane & 15)) * HIDDEN + hoff + kk0;
            #pragma unroll
            for (int j = 0; j < 16; ++j) {
                a.u[j]  = qs[(lane & 15) * 72 + kk0 + ka_idx(j, lane)];
                bf.u[j] = kp[kb_idx(j, lane)];
            }
            acc = __builtin_amdgcn_wmma_f32_16x16x32_bf16(false, a.v, false, bf.v,
                                                          (short)0, acc, false, false);
        }
        #pragma unroll
        for (int i = 0; i < 8; ++i) {
            int m = i + ((lane >> 4) << 3);
            sc[m * SEQ + key0 + (lane & 15)] = acc[i] * 0.125f;   // 1/sqrt(64)
        }
    }
    __syncthreads();

    // ---- Phase 2: per-row top-204 threshold (bisection on uint order-keys)
    //      + softmax stats. Wave w owns rows 4w..4w+3. ----
    for (int r = 0; r < 4; ++r) {
        int row = wave * 4 + r;
        const float* srow = sc + row * SEQ;

        float mx = -3.0e38f;
        for (int i = lane; i < SEQ; i += 32) mx = fmaxf(mx, srow[i]);
        #pragma unroll
        for (int s = 16; s > 0; s >>= 1) mx = fmaxf(mx, __shfl_xor(mx, s, 32));

        unsigned lo = 0u, hi = 0xFFFFFFFFu;     // largest t with count(>=t) >= KK_TOP
        for (int it = 0; it < 32; ++it) {
            unsigned mid = (unsigned)(((unsigned long long)lo + hi + 1ull) >> 1);
            int c = 0;
            for (int i = lane; i < SEQ; i += 32) c += (ordkey(srow[i]) >= mid);
            #pragma unroll
            for (int s = 16; s > 0; s >>= 1) c += __shfl_xor(c, s, 32);
            if (c >= KK_TOP) lo = mid; else hi = mid - 1u;
        }

        float sum = 0.f;
        for (int i = lane; i < SEQ; i += 32) {
            float v = srow[i];
            if (ordkey(v) >= lo) sum += __expf(v - mx);
        }
        #pragma unroll
        for (int s = 16; s > 0; s >>= 1) sum += __shfl_xor(sum, s, 32);

        if (lane == 0) { thrK[row] = lo; rowMax[row] = mx; rowSum[row] = sum; }
    }
    __syncthreads();

    // ---- Phase 3: O = P x V.  Each wave owns a 16-wide head-dim slice. ----
    const int dh0 = wave * 16;
    v8f acc = {};
    for (int key0 = 0; key0 < SEQ; key0 += 32) {
        BFrag a, bf;
        #pragma unroll
        for (int j = 0; j < 16; ++j) {
            int m  = lane & 15;
            float v = sc[m * SEQ + key0 + ka_idx(j, lane)];
            float p = (ordkey(v) >= thrK[m]) ? __expf(v - rowMax[m]) / rowSum[m] : 0.f;
            a.u[j] = f2bf(p);
            bf.u[j] = V[(base + key0 + kb_idx(j, lane)) * HIDDEN + hoff + dh0 + (lane & 15)];
        }
        acc = __builtin_amdgcn_wmma_f32_16x16x32_bf16(false, a.v, false, bf.v,
                                                      (short)0, acc, false, false);
    }
    #pragma unroll
    for (int i = 0; i < 8; ++i) {
        int m = q0 + i + ((lane >> 4) << 3);
        O[(base + m) * HIDDEN + hoff + dh0 + (lane & 15)] = f2bf(acc[i]);
    }
}

// ---------------------------------------------------------------------------
// Host-side orchestration
// ---------------------------------------------------------------------------
extern "C" void kernel_launch(void* const* d_in, const int* in_sizes, int n_in,
                              void* d_out, int out_size, void* d_ws, size_t ws_size,
                              hipStream_t stream) {
    const float* x  = (const float*)d_in[0];
    const float* Wq = (const float*)d_in[1];
    const float* bq = (const float*)d_in[2];
    const float* Wk = (const float*)d_in[3];
    const float* bk = (const float*)d_in[4];
    const float* Wv = (const float*)d_in[5];
    const float* bv = (const float*)d_in[6];
    const float* Wo = (const float*)d_in[7];
    const float* bo = (const float*)d_in[8];

    char* ws = (char*)d_ws;
    const size_t WB = (size_t)HIDDEN * HIDDEN * sizeof(unsigned short);   // 1.18 MB
    const size_t XB = (size_t)MTOT * HIDDEN * sizeof(unsigned short);     // 6.29 MB
    unsigned short* wqb = (unsigned short*)(ws);
    unsigned short* wkb = (unsigned short*)(ws + WB);
    unsigned short* wvb = (unsigned short*)(ws + 2 * WB);
    unsigned short* wob = (unsigned short*)(ws + 3 * WB);
    unsigned short* xb  = (unsigned short*)(ws + 4 * WB);
    unsigned short* Qb  = (unsigned short*)(ws + 4 * WB + XB);
    unsigned short* Kb  = (unsigned short*)(ws + 4 * WB + 2 * XB);
    unsigned short* Vb  = (unsigned short*)(ws + 4 * WB + 3 * XB);
    unsigned short* AOb = (unsigned short*)(ws + 4 * WB + 4 * XB);        // end ~36.2 MB

    const int nw = HIDDEN * HIDDEN;
    const int nx = MTOT * HIDDEN;
    cvt_f32_bf16<<<(nw + 255) / 256, 256, 0, stream>>>(Wq, wqb, nw);
    cvt_f32_bf16<<<(nw + 255) / 256, 256, 0, stream>>>(Wk, wkb, nw);
    cvt_f32_bf16<<<(nw + 255) / 256, 256, 0, stream>>>(Wv, wvb, nw);
    cvt_f32_bf16<<<(nw + 255) / 256, 256, 0, stream>>>(Wo, wob, nw);
    cvt_f32_bf16<<<(nx + 255) / 256, 256, 0, stream>>>(x,  xb,  nx);

    dim3 gg(HIDDEN / 64, MTOT / 128);               // 12 x 32 workgroups
    const size_t gsmem = 6144 + 128 * 48 * 2;       // Bs + As = 18 KB
    gemm_nt_bf16<true><<<gg, 256, gsmem, stream>>>(xb, wqb, bq, Qb, MTOT, HIDDEN, HIDDEN);
    gemm_nt_bf16<true><<<gg, 256, gsmem, stream>>>(xb, wkb, bk, Kb, MTOT, HIDDEN, HIDDEN);
    gemm_nt_bf16<true><<<gg, 256, gsmem, stream>>>(xb, wvb, bv, Vb, MTOT, HIDDEN, HIDDEN);

    const size_t asmem = (size_t)16 * SEQ * 4 + 16 * 72 * 2 + 16 * 4 * 3; // ~133.5 KB
    dim3 ag(SEQ / 16, HEADS, BATCH);                // 128 x 12 x 2 workgroups
    sparse_attn<<<ag, 128, asmem, stream>>>(Qb, Kb, Vb, AOb);

    gemm_nt_bf16<false><<<gg, 256, gsmem, stream>>>(AOb, wob, bo, d_out, MTOT, HIDDEN, HIDDEN);
}
